// ReuploadingQuantumAttention_79723182949046
// MI455X (gfx1250) — matrix operand order repaired and checked by reading, not compiled, not run on hardware
//
#include <hip/hip_runtime.h>
#include <hip/hip_bf16.h>
#include <math.h>

// CDNA5 / gfx1250 vector types for WMMA + native vectors for NT loads
typedef __attribute__((ext_vector_type(16))) _Float16 v16h;
typedef __attribute__((ext_vector_type(8)))  float    v8f;
typedef __attribute__((ext_vector_type(4)))  float    v4f;

#define NQ      4
#define DIM     16
#define NLAYERS 3

__device__ __forceinline__ float2 cmul(float2 a, float2 b) {
    return make_float2(a.x * b.x - a.y * b.y, a.x * b.y + a.y * b.x);
}

// edge_attr: 410MB single-pass stream (>> 192MB L2), bypass cache retention
__device__ __forceinline__ v4f ldg_nt(const float* p) {
    return __builtin_nontemporal_load(reinterpret_cast<const v4f*>(p));
}

// Branchless tanh: (e^{2x}-1)/(e^{2x}+1); clamp keeps e^{2x} finite.
// ~6 straight-line VALU ops vs libm tanhf's exec-mask branches.
__device__ __forceinline__ float fast_tanh(float x) {
    x = fminf(fmaxf(x, -9.f), 9.f);
    float e = __expf(2.f * x);
    return (e - 1.f) / (e + 1.f);
}

__global__ __launch_bounds__(256)
void rqa_kernel(const float* __restrict__ edge,  // (n, 128)
                const float* __restrict__ Wp,    // (4, 128)
                const float* __restrict__ bp,    // (4,)
                const float* __restrict__ qw,    // (3, 4, 3)
                float* __restrict__ out,         // (n, 1)
                int nRows) {
    __shared__ float s_rot[12][8];    // 3 layers x 4 wires: 2x2 complex Rot gates
    __shared__ float s_ang[256][4];   // per-sample embedding angles

    const int tid  = threadIdx.x;
    const int wid  = tid >> 5;        // wave id (wave32)
    const int lane = tid & 31;

    // ---- Phase 0: constant Rot = RZ(omega) RY(theta) RZ(phi) matrices, once per block
    if (tid < 12) {
        const float* g = qw + tid * 3;
        float phi = g[0], th = g[1], om = g[2];
        float st, ct; __sincosf(0.5f * th, &st, &ct);
        float sp, cp; __sincosf(0.5f * (phi + om), &sp, &cp);
        float sm, cm; __sincosf(0.5f * (phi - om), &sm, &cm);
        s_rot[tid][0] =  cp * ct;  s_rot[tid][1] = -sp * ct;   // U00 = e^{-i(phi+om)/2} c
        s_rot[tid][2] = -cm * st;  s_rot[tid][3] = -sm * st;   // U01 = -e^{+i(phi-om)/2} s
        s_rot[tid][4] =  cm * st;  s_rot[tid][5] = -sm * st;   // U10 = e^{-i(phi-om)/2} s
        s_rot[tid][6] =  cp * ct;  s_rot[tid][7] =  sp * ct;   // U11 = e^{+i(phi+om)/2} c
    }
    __syncthreads();

    const int col = lane & 15;   // WMMA B/C column (qubit index for col < 4)
    const int hi  = lane >> 4;   // which half of the K striping this lane holds

    // ---- Phase 1a: B fragments (W_proj) — tile-invariant, hoisted, branch-free.
    // B is 32x16 f16 per chunk: lane n holds B[k][n] = W_proj[n][k];
    // lanes 0-15 carry K 0-7 & 16-23, lanes 16-31 carry K 8-15 & 24-31 of each chunk.
    // Columns >= 4 are zeroed by mask (keeps EXEC uniform, no exec-mask dance).
    const float wmask = (col < NQ) ? 1.f : 0.f;
    const float* wrow = Wp + (col & 3) * 128;   // safe row, result masked
    v16h bfrag[4];
    #pragma unroll
    for (int kb = 0; kb < 4; ++kb) {
        const float* bptr = wrow + kb * 32 + hi * 8;
        v4f b0 = *reinterpret_cast<const v4f*>(bptr + 0);
        v4f b1 = *reinterpret_cast<const v4f*>(bptr + 4);
        v4f b2 = *reinterpret_cast<const v4f*>(bptr + 16);
        v4f b3 = *reinterpret_cast<const v4f*>(bptr + 20);
        v16h b;
        b[0]=(_Float16)(wmask*b0.x);  b[1]=(_Float16)(wmask*b0.y);
        b[2]=(_Float16)(wmask*b0.z);  b[3]=(_Float16)(wmask*b0.w);
        b[4]=(_Float16)(wmask*b1.x);  b[5]=(_Float16)(wmask*b1.y);
        b[6]=(_Float16)(wmask*b1.z);  b[7]=(_Float16)(wmask*b1.w);
        b[8]=(_Float16)(wmask*b2.x);  b[9]=(_Float16)(wmask*b2.y);
        b[10]=(_Float16)(wmask*b2.z); b[11]=(_Float16)(wmask*b2.w);
        b[12]=(_Float16)(wmask*b3.x); b[13]=(_Float16)(wmask*b3.y);
        b[14]=(_Float16)(wmask*b3.z); b[15]=(_Float16)(wmask*b3.w);
        bfrag[kb] = b;
    }
    const float bias = (col < NQ) ? bp[col] : 0.f;

    // ---- Phase 1b: projection GEMM on the matrix pipe, two 16-sample tiles per wave
    #pragma unroll
    for (int t = 0; t < 2; ++t) {
        int tileBase = blockIdx.x * 256 + wid * 32 + t * 16;
        int r = tileBase + col;                 // A-matrix row (sample) for this lane
        if (r >= nRows) r = nRows - 1;          // clamp (grid divides exactly; safety)
        const float* arow = edge + (size_t)r * 128;

        v8f acc;                                // C init = bias broadcast down the column
        #pragma unroll
        for (int j = 0; j < 8; ++j) acc[j] = bias;

        #pragma unroll
        for (int kb = 0; kb < 4; ++kb) {        // K = 128 in chunks of 32
            // A fragment (16x32 f16 layout): lanes 0-15 hold K 0-7 & 16-23,
            // lanes 16-31 hold K 8-15 & 24-31 of this chunk — 32B-aligned, NT b128 loads.
            const float* aptr = arow + kb * 32 + hi * 8;
            v4f a0 = ldg_nt(aptr + 0);
            v4f a1 = ldg_nt(aptr + 4);
            v4f a2 = ldg_nt(aptr + 16);
            v4f a3 = ldg_nt(aptr + 20);
            v16h a;
            a[0]=(_Float16)a0.x;  a[1]=(_Float16)a0.y;  a[2]=(_Float16)a0.z;  a[3]=(_Float16)a0.w;
            a[4]=(_Float16)a1.x;  a[5]=(_Float16)a1.y;  a[6]=(_Float16)a1.z;  a[7]=(_Float16)a1.w;
            a[8]=(_Float16)a2.x;  a[9]=(_Float16)a2.y;  a[10]=(_Float16)a2.z; a[11]=(_Float16)a2.w;
            a[12]=(_Float16)a3.x; a[13]=(_Float16)a3.y; a[14]=(_Float16)a3.z; a[15]=(_Float16)a3.w;

            acc = __builtin_amdgcn_wmma_f32_16x16x32_f16(
                /*neg_a=*/false, a, /*neg_b=*/false, bfrag[kb],
                /*c_mod=*/(short)0, acc, /*reuse_a=*/false, /*reuse_b=*/false);
        }

        // angles = tanh(D) * pi — computed in all lanes (uniform EXEC, branchless tanh);
        // only the 8 lanes with col<4 commit to LDS.
        // D layout: lanes<16 hold rows m=j, lanes>=16 hold rows m=8+j of column `col`.
        #pragma unroll
        for (int j = 0; j < 8; ++j) {
            float ang = fast_tanh(acc[j]) * 3.14159265358979f;
            if (col < NQ)
                s_ang[wid * 32 + t * 16 + hi * 8 + j][col] = ang;
        }
    }
    __syncthreads();

    // ---- Phase 2: one sample per lane, 16 complex amplitudes in registers
    const int localS = wid * 32 + lane;
    const int gid    = blockIdx.x * 256 + localS;

    float cw[NQ], sw[NQ];
    #pragma unroll
    for (int q = 0; q < NQ; ++q) {
        float th = s_ang[localS][q];            // same angles re-uploaded each layer
        __sincosf(0.5f * th, &sw[q], &cw[q]);
    }

    float2 amp[DIM];
    #pragma unroll
    for (int i = 0; i < DIM; ++i) amp[i] = make_float2(0.f, 0.f);
    amp[0].x = 1.f;

    #pragma unroll
    for (int l = 0; l < NLAYERS; ++l) {
        // AngleEmbedding: RY(theta_w) on wire w (wire 0 = MSB -> bit mask 8>>w)
        #pragma unroll
        for (int w = 0; w < NQ; ++w) {
            const int mask = 8 >> w;
            const float c = cw[w], s = sw[w];
            #pragma unroll
            for (int i = 0; i < DIM; ++i) {
                if (i & mask) continue;
                float2 a0 = amp[i], a1 = amp[i | mask];
                amp[i]        = make_float2(c * a0.x - s * a1.x, c * a0.y - s * a1.y);
                amp[i | mask] = make_float2(s * a0.x + c * a1.x, s * a0.y + c * a1.y);
            }
        }
        // StronglyEntangling Rot gates (constant, from LDS)
        #pragma unroll
        for (int w = 0; w < NQ; ++w) {
            const float* R = s_rot[l * 4 + w];
            float2 U00 = make_float2(R[0], R[1]);
            float2 U01 = make_float2(R[2], R[3]);
            float2 U10 = make_float2(R[4], R[5]);
            float2 U11 = make_float2(R[6], R[7]);
            const int mask = 8 >> w;
            #pragma unroll
            for (int i = 0; i < DIM; ++i) {
                if (i & mask) continue;
                float2 a0 = amp[i], a1 = amp[i | mask];
                float2 r0 = cmul(U00, a0), r0b = cmul(U01, a1);
                float2 r1 = cmul(U10, a0), r1b = cmul(U11, a1);
                amp[i]        = make_float2(r0.x + r0b.x, r0.y + r0b.y);
                amp[i | mask] = make_float2(r1.x + r1b.x, r1.y + r1b.y);
            }
        }
        // CNOT ring: control w -> target (w+1)%4 ; pure register swaps
        #pragma unroll
        for (int w = 0; w < NQ; ++w) {
            const int cm = 8 >> w;
            const int tm = 8 >> ((w + 1) & 3);
            #pragma unroll
            for (int i = 0; i < DIM; ++i) {
                if ((i & cm) && !(i & tm)) {
                    float2 tmp   = amp[i];
                    amp[i]       = amp[i | tm];
                    amp[i | tm]  = tmp;
                }
            }
        }
    }

    // <Z_0> = P(q0=0) - P(q0=1); q0 is the MSB of the state index
    float z = 0.f;
    #pragma unroll
    for (int i = 0; i < DIM; ++i) {
        float p = amp[i].x * amp[i].x + amp[i].y * amp[i].y;
        z += (i < 8) ? p : -p;
    }
    float sig = 1.f / (1.f + __expf(-z));
    if (gid < nRows) __builtin_nontemporal_store(sig, out + gid);
}

extern "C" void kernel_launch(void* const* d_in, const int* in_sizes, int n_in,
                              void* d_out, int out_size, void* d_ws, size_t ws_size,
                              hipStream_t stream) {
    const float* edge = (const float*)d_in[0];  // (N, 128) f32
    const float* Wp   = (const float*)d_in[1];  // (4, 128) f32
    const float* bp   = (const float*)d_in[2];  // (4,)     f32
    const float* qw   = (const float*)d_in[3];  // (3,4,3)  f32
    float* out = (float*)d_out;                 // (N, 1)   f32

    int n = in_sizes[0] / 128;                  // 800000
    int grid = (n + 255) / 256;                 // 3125 blocks of 8 waves
    rqa_kernel<<<grid, 256, 0, stream>>>(edge, Wp, bp, qw, out, n);
}
